// NodeModel_28346784153766
// MI455X (gfx1250) — matrix operand order repaired and checked by reading, not compiled, not run on hardware
//
#include <hip/hip_runtime.h>
#include <hip/hip_bf16.h>

// ---------------------------------------------------------------------------
// Problem constants (from the reference)
// ---------------------------------------------------------------------------
#define NNODES 50000
#define NEDGES 800000
#define NODE_DIM 128
#define EDGE_DIM 128
#define IN_DIM   256   // NODE_DIM + EDGE_DIM
#define HID_DIM  256
#define OUT_DIM  128
#define NUM_TILES (NNODES / 16)    // 3125, exact

typedef _Float16 v16h __attribute__((ext_vector_type(16)));
typedef _Float16 v8h  __attribute__((ext_vector_type(8)));
typedef float    v8f  __attribute__((ext_vector_type(8)));

// ---------------------------------------------------------------------------
// Kernel 1: zero the aggregation buffer (float4 stores, pure BW)
// ---------------------------------------------------------------------------
__global__ __launch_bounds__(256) void zero_agg_kernel(float4* agg4, int n4) {
    int i = blockIdx.x * blockDim.x + threadIdx.x;
    if (i < n4) {
        float4 z; z.x = 0.f; z.y = 0.f; z.z = 0.f; z.w = 0.f;
        agg4[i] = z;
    }
}

// ---------------------------------------------------------------------------
// Kernel 2: convert W1 / W2 to f16 (contiguous in ws: W1h then W2h)
// ---------------------------------------------------------------------------
__global__ __launch_bounds__(256) void prep_weights_kernel(const float* __restrict__ W1,
                                                           const float* __restrict__ W2,
                                                           _Float16* __restrict__ W1h,
                                                           _Float16* __restrict__ W2h) {
    int i = blockIdx.x * blockDim.x + threadIdx.x;
    if (i < HID_DIM * IN_DIM)  W1h[i] = (_Float16)W1[i];
    if (i < OUT_DIM * HID_DIM) W2h[i] = (_Float16)W2[i];
}

// ---------------------------------------------------------------------------
// Kernel 3: scatter-add edge_attr into agg by destination node.
// One thread per (edge, 4-float chunk): b128 load + 4 L2 f32 atomics.
// ---------------------------------------------------------------------------
__global__ __launch_bounds__(256) void scatter_edges_kernel(const float* __restrict__ edge_attr,
                                                            const long long* __restrict__ edge_index,
                                                            float* __restrict__ agg) {
    long long tid = (long long)blockIdx.x * blockDim.x + threadIdx.x;
    int e = (int)(tid >> 5);          // 32 chunks of 4 floats per edge
    int c = ((int)tid & 31) << 2;     // starting float within the row
    if (e >= NEDGES) return;
    int col = (int)edge_index[(long long)NEDGES + e];   // row 1 of edge_index
    const float4 v = *(const float4*)(edge_attr + (size_t)e * EDGE_DIM + c);
    float* dst = agg + (size_t)col * EDGE_DIM + c;
    __hip_atomic_fetch_add(dst + 0, v.x, __ATOMIC_RELAXED, __HIP_MEMORY_SCOPE_AGENT);
    __hip_atomic_fetch_add(dst + 1, v.y, __ATOMIC_RELAXED, __HIP_MEMORY_SCOPE_AGENT);
    __hip_atomic_fetch_add(dst + 2, v.z, __ATOMIC_RELAXED, __HIP_MEMORY_SCOPE_AGENT);
    __hip_atomic_fetch_add(dst + 3, v.w, __ATOMIC_RELAXED, __HIP_MEMORY_SCOPE_AGENT);
}

// ---------------------------------------------------------------------------
// Kernel 4: fused 2-layer MLP via v_wmma_f32_16x16x32_f16.
//   - Weights (W1h|W2h = 192KB f16) staged into LDS once per block.
//   - 8 waves/block; each wave grid-strides over 16-node tiles.
//   - A fragments cached in registers (8 x v16h); A LDS region is then dead
//     and reused for the H (hidden) tile. 8KB activation LDS per wave.
//   - Total LDS: 192KB weights + 64KB activations = 256KB (<= 320KB/WGP).
// ---------------------------------------------------------------------------
#define NWAVES 8
#define W_HALVES (HID_DIM * IN_DIM + OUT_DIM * HID_DIM)   // 98304 halves, 192KB
#define ACT_HALVES (16 * 256)                             // 4096 halves, 8KB

__device__ __forceinline__ v16h frag_split(const _Float16* p) {
    // A-matrix 16-bit layout: halves [0..7] at p, halves [8..15] at p+16
    union { uint4 q[2]; v16h h; } u;
    u.q[0] = *(const uint4*)(p);
    u.q[1] = *(const uint4*)(p + 16);
    return u.h;
}

__device__ __forceinline__ v16h frag_contig(const _Float16* p) {
    // Contiguous k order (k = 16h + s): 16 consecutive K halves, used for both A and B
    union { uint4 q[2]; v16h h; } u;
    u.q[0] = *(const uint4*)(p);
    u.q[1] = *(const uint4*)(p + 8);
    return u.h;
}

__global__ __launch_bounds__(32 * NWAVES) void fused_mlp_kernel(
        const float* __restrict__ x,
        const float* __restrict__ agg,
        const _Float16* __restrict__ Wh,     // W1h followed by W2h, contiguous
        const float* __restrict__ b1,
        const float* __restrict__ b2,
        float* __restrict__ out) {

    __shared__ _Float16 wlds[W_HALVES];               // 192KB: W1 then W2
    __shared__ _Float16 act[NWAVES][ACT_HALVES];      // 64KB: per-wave A/H tile

    // ---- Stage all weights into LDS (cooperative, b128 copies) ------------
    {
        const uint4* src = (const uint4*)Wh;
        uint4* dst = (uint4*)wlds;
        for (int i = threadIdx.x; i < W_HALVES / 8; i += 32 * NWAVES)
            dst[i] = src[i];
    }
    __syncthreads();

    const int wave   = threadIdx.x >> 5;
    const int lane   = threadIdx.x & 31;
    const int hiHalf = lane >> 4;          // 0: lanes 0-15, 1: lanes 16-31
    const int lane15 = lane & 15;
    const int mbase  = hiHalf * 8;         // C/D rows handled by this lane

    _Float16* T = &act[wave][0];           // A tile, later reused as H tile
    const _Float16* W1L = wlds;
    const _Float16* W2L = wlds + HID_DIM * IN_DIM;

    for (int tile = blockIdx.x * NWAVES + wave; tile < NUM_TILES;
         tile += gridDim.x * NWAVES) {
        const int node_base = tile * 16;

        // ---- Stage A tile: row = it, lane covers 8 consecutive features ---
        // lanes 0..15 pull from x, lanes 16..31 from agg. f32 -> f16.
        for (int it = 0; it < 16; ++it) {
            const int col = lane * 8;      // 0..248
            const float* src = (col < NODE_DIM)
                ? (x   + (size_t)(node_base + it) * NODE_DIM + col)
                : (agg + (size_t)(node_base + it) * EDGE_DIM + (col - NODE_DIM));
            float4 a = *(const float4*)(src);
            float4 b = *(const float4*)(src + 4);
            v8h hv;
            hv[0] = (_Float16)a.x; hv[1] = (_Float16)a.y;
            hv[2] = (_Float16)a.z; hv[3] = (_Float16)a.w;
            hv[4] = (_Float16)b.x; hv[5] = (_Float16)b.y;
            hv[6] = (_Float16)b.z; hv[7] = (_Float16)b.w;
            *(v8h*)(T + it * IN_DIM + col) = hv;       // ds_store_b128
        }
        // Same-wave DS ops are in-order: reads below observe these stores.

        // ---- Load all 8 A fragments into registers (A LDS now dead) -------
        // lane<16: row lane15, K {0..15}; lane>=16: K {16..31}
        v16h af[8];
        #pragma unroll
        for (int k = 0; k < 8; ++k)
            af[k] = frag_contig(T + lane15 * IN_DIM + k * 32 + hiHalf * 16);

        // ---- GEMM1: H = relu(A @ W1^T + b1); H overwrites the A region ----
        #pragma unroll 2
        for (int nt = 0; nt < 16; ++nt) {
            const float bv = b1[nt * 16 + lane15];
            v8f acc;
            #pragma unroll
            for (int v = 0; v < 8; ++v) acc[v] = bv;
            #pragma unroll
            for (int k = 0; k < 8; ++k) {
                const int kbase = k * 32 + hiHalf * 16;
                v16h bf = frag_contig(W1L + (nt * 16 + lane15) * IN_DIM + kbase);
                acc = __builtin_amdgcn_wmma_f32_16x16x32_f16(
                          false, af[k], false, bf, (short)0, acc, false, false);
            }
            // ReLU + stash H as f16 (C layout: lane -> col, vgpr -> row)
            const int nc = nt * 16 + lane15;
            #pragma unroll
            for (int v = 0; v < 8; ++v) {
                float r = acc[v] > 0.f ? acc[v] : 0.f;
                T[(mbase + v) * HID_DIM + nc] = (_Float16)r;
            }
        }

        // ---- Load all 8 H fragments into registers ------------------------
        v16h ah[8];
        #pragma unroll
        for (int k = 0; k < 8; ++k)
            ah[k] = frag_contig(T + lane15 * HID_DIM + k * 32 + hiHalf * 16);

        // ---- GEMM2: out = H @ W2^T + b2 -----------------------------------
        #pragma unroll 2
        for (int mt = 0; mt < 8; ++mt) {
            const float bv = b2[mt * 16 + lane15];
            v8f acc;
            #pragma unroll
            for (int v = 0; v < 8; ++v) acc[v] = bv;
            #pragma unroll
            for (int k = 0; k < 8; ++k) {
                const int kbase = k * 32 + hiHalf * 16;
                v16h bf = frag_contig(W2L + (mt * 16 + lane15) * HID_DIM + kbase);
                acc = __builtin_amdgcn_wmma_f32_16x16x32_f16(
                          false, ah[k], false, bf, (short)0, acc, false, false);
            }
            const int nc = mt * 16 + lane15;
            #pragma unroll
            for (int v = 0; v < 8; ++v)
                out[(size_t)(node_base + mbase + v) * OUT_DIM + nc] = acc[v];
        }
    }
}

// ---------------------------------------------------------------------------
// Host launcher
// ---------------------------------------------------------------------------
extern "C" void kernel_launch(void* const* d_in, const int* in_sizes, int n_in,
                              void* d_out, int out_size, void* d_ws, size_t ws_size,
                              hipStream_t stream) {
    (void)in_sizes; (void)n_in; (void)out_size; (void)ws_size;

    // setup_inputs() order: x, edge_index, edge_attr, u, batch, W1, b1, W2, b2
    const float*     x          = (const float*)d_in[0];
    const long long* edge_index = (const long long*)d_in[1];   // int64 [2, E]
    const float*     edge_attr  = (const float*)d_in[2];
    const float*     W1         = (const float*)d_in[5];
    const float*     b1         = (const float*)d_in[6];
    const float*     W2         = (const float*)d_in[7];
    const float*     b2         = (const float*)d_in[8];
    float*           out        = (float*)d_out;

    // Workspace layout: [agg f32 N*128][W1h f16][W2h f16] (W1h,W2h contiguous)
    char* ws = (char*)d_ws;
    float*    agg = (float*)ws;                                   // 25,600,000 B
    _Float16* W1h = (_Float16*)(ws + (size_t)NNODES * EDGE_DIM * sizeof(float));
    _Float16* W2h = W1h + (size_t)HID_DIM * IN_DIM;

    // 1) zero agg
    {
        int n4 = NNODES * EDGE_DIM / 4;
        zero_agg_kernel<<<(n4 + 255) / 256, 256, 0, stream>>>((float4*)agg, n4);
    }
    // 2) f16 weights
    {
        int n = HID_DIM * IN_DIM;   // 65536 covers W2 (32768) too
        prep_weights_kernel<<<(n + 255) / 256, 256, 0, stream>>>(W1, W2, W1h, W2h);
    }
    // 3) scatter-add (one thread per edge x float4 chunk)
    {
        long long total = (long long)NEDGES * 32;
        int blocks = (int)((total + 255) / 256);
        scatter_edges_kernel<<<blocks, 256, 0, stream>>>(edge_attr, edge_index, agg);
    }
    // 4) fused WMMA MLP: 128 blocks x 8 waves grid-stride over 3125 tiles
    {
        fused_mlp_kernel<<<128, 32 * NWAVES, 0, stream>>>(
            x, agg, W1h, b1, b2, out);
    }
}
